// LACEModel_84713934946426
// MI455X (gfx1250) — compile-verified, matrix operand order checked by reading, not run on hardware
//
#include <hip/hip_runtime.h>

typedef _Float16 h16;
typedef __attribute__((ext_vector_type(16))) _Float16 v16h;
typedef __attribute__((ext_vector_type(8)))  _Float16 h16x8;
typedef __attribute__((ext_vector_type(8)))  float    v8f;

#define N_DOCS   4
#define SEQL     1024
#define HIDD     768
#define NHEADS   12
#define NENT     48
#define NMEN     4
#define NPAIRS   1000
#define NLAB     97
#define EMBD     768
#define BLKD     64
#define GHIDD    500
#define NEDGES   897
#define TOTPAIR  (N_DOCS*NPAIRS)   // 4000
#define PAIRPAD  4096              // rows padded for guard-free A loads
#define KBIL     (EMBD*BLKD)       // 49152

// load 16 f16 laid out as [k..k+7],[k+16..k+23]  -> two global_load_b128
__device__ __forceinline__ v16h ld_frag(const h16* __restrict__ p) {
  union { v16h v; struct { h16x8 lo, hi; } s; } u;
  u.s.lo = *(const h16x8*)p;
  u.s.hi = *(const h16x8*)(p + 16);
  return u.v;
}

// ---------------------------------------------------------------------------
// Guard-free WMMA GEMM:  C[M,N] = epi( A[M,K] * BT[N,K]^T )
// A row-major [Mp][lda], BT row-major [Np][ldb] (i.e. B transposed), both f16,
// zero-padded so fragment loads need no bounds checks. 128 thr = 4 waves,
// block tile 64x64, wave tile 32x32 (2x2 wmma). No LDS, no barriers.
// epi: 0=none, 1=+bias, 2=tanh(x+bias), 3=C+=x. Batched via blockIdx.z.
// ---------------------------------------------------------------------------
__global__ __launch_bounds__(128)
void k_gemm(const h16* __restrict__ A, const h16* __restrict__ BT,
            float* __restrict__ C, const float* __restrict__ bias,
            int M, int N, int Kp, int lda, int ldb, int ldc,
            long long strideA, long long strideB, long long strideC, int epi)
{
  A  += (long long)blockIdx.z * strideA;
  BT += (long long)blockIdx.z * strideB;
  C  += (long long)blockIdx.z * strideC;
  const int tid  = threadIdx.x;
  const int lane = tid & 31;
  const int wave = tid >> 5;
  const int wr   = (wave >> 1) * 32;
  const int wc   = (wave & 1) * 32;
  const int m0   = blockIdx.y * 64;
  const int n0   = blockIdx.x * 64;
  const int lh   = lane & 15;
  const int ks   = (lane >> 4) * 8;

  const h16* a0 = A  + (long long)(m0 + wr + lh) * lda + ks;
  const h16* a1 = a0 + (long long)16 * lda;
  const h16* b0 = BT + (long long)(n0 + wc + lh) * ldb + ks;
  const h16* b1 = b0 + (long long)16 * ldb;

  v8f acc[2][2];
  for (int i = 0; i < 2; i++)
    for (int j = 0; j < 2; j++)
      for (int e = 0; e < 8; e++) acc[i][j][e] = 0.f;

  for (int k0 = 0; k0 < Kp; k0 += 32) {
    v16h af0 = ld_frag(a0 + k0);
    v16h af1 = ld_frag(a1 + k0);
    v16h bf0 = ld_frag(b0 + k0);
    v16h bf1 = ld_frag(b1 + k0);
    acc[0][0] = __builtin_amdgcn_wmma_f32_16x16x32_f16(false, af0, false, bf0, (short)0, acc[0][0], false, false);
    acc[0][1] = __builtin_amdgcn_wmma_f32_16x16x32_f16(false, af0, false, bf1, (short)0, acc[0][1], false, false);
    acc[1][0] = __builtin_amdgcn_wmma_f32_16x16x32_f16(false, af1, false, bf0, (short)0, acc[1][0], false, false);
    acc[1][1] = __builtin_amdgcn_wmma_f32_16x16x32_f16(false, af1, false, bf1, (short)0, acc[1][1], false, false);
  }

  const int mo = (lane >> 4) * 8;
  for (int i = 0; i < 2; i++) {
    int rowbase = m0 + wr + i * 16 + mo;
    for (int j = 0; j < 2; j++) {
      int n = n0 + wc + j * 16 + lh;
      if (n >= N) continue;
      float bv = (epi == 1 || epi == 2) ? bias[n] : 0.f;
      for (int r = 0; r < 8; r++) {
        int row = rowbase + r;
        if (row >= M) break;
        float v = acc[i][j][r] + bv;
        if (epi == 2) v = tanhf(v);
        long long o = (long long)row * ldc + n;
        if (epi == 3) v += C[o];
        C[o] = v;
      }
    }
  }
}

// ---------------------------------------------------------------------------
// Bilinear head: out[p,l] = sum_{k,i,j} h2[p,k*64+i]*t2[p,k*64+j]*W[k*4096+i*64+j,l]
// A fragment built in registers (no LDS/barriers): lane owns row m0+lh.
// WbT is bil_W[:49152] transposed+padded: [112][49152]. 224 thr = 7 N-tiles.
// ---------------------------------------------------------------------------
__global__ __launch_bounds__(224)
void k_bilinear(const float* __restrict__ h2, const float* __restrict__ t2,
                const h16* __restrict__ WbT, const float* __restrict__ bb,
                float* __restrict__ out)
{
  const int tid  = threadIdx.x;
  const int lane = tid & 31;
  const int wave = tid >> 5;           // 0..6
  const int m0   = blockIdx.x * 16;
  const int n0   = wave * 16;
  const int lh   = lane & 15;
  const int ks   = (lane >> 4) * 8;

  const float* h2r = h2 + (long long)(m0 + lh) * EMBD;
  const float* t2r = t2 + (long long)(m0 + lh) * EMBD;
  const h16*  brow = WbT + (long long)(n0 + lh) * KBIL + ks;

  v8f acc;
  for (int e = 0; e < 8; e++) acc[e] = 0.f;

  for (int kk0 = 0; kk0 < KBIL; kk0 += 32) {
    const int blk = kk0 >> 12;
    const int ii  = (kk0 >> 6) & 63;
    const int j0  = kk0 & 63;
    float hv = h2r[blk * 64 + ii];
    const float4* tp0 = (const float4*)(t2r + blk * 64 + j0 + ks);
    const float4* tp1 = (const float4*)(t2r + blk * 64 + j0 + ks + 16);
    float4 t0a = tp0[0], t0b = tp0[1];
    float4 t1a = tp1[0], t1b = tp1[1];
    v16h af;
    af[0]  = (h16)(hv * t0a.x); af[1]  = (h16)(hv * t0a.y);
    af[2]  = (h16)(hv * t0a.z); af[3]  = (h16)(hv * t0a.w);
    af[4]  = (h16)(hv * t0b.x); af[5]  = (h16)(hv * t0b.y);
    af[6]  = (h16)(hv * t0b.z); af[7]  = (h16)(hv * t0b.w);
    af[8]  = (h16)(hv * t1a.x); af[9]  = (h16)(hv * t1a.y);
    af[10] = (h16)(hv * t1a.z); af[11] = (h16)(hv * t1a.w);
    af[12] = (h16)(hv * t1b.x); af[13] = (h16)(hv * t1b.y);
    af[14] = (h16)(hv * t1b.z); af[15] = (h16)(hv * t1b.w);
    v16h bf = ld_frag(brow + kk0);
    acc = __builtin_amdgcn_wmma_f32_16x16x32_f16(false, af, false, bf, (short)0, acc, false, false);
  }

  int n = n0 + lh;
  if (n < NLAB) {
    int rowbase = m0 + ((lane >> 4) * 8);
    float bv = bb[n];
    for (int r = 0; r < 8; r++)
      out[(long long)(rowbase + r) * NLAB + n] = acc[r] + bv;
  }
}

// ---------------------------------------------------------------------------
// Small helper kernels
// ---------------------------------------------------------------------------
__global__ void k_zero_u32(unsigned int* p, long long n) {
  long long t = (long long)blockIdx.x * blockDim.x + threadIdx.x;
  if (t < n) p[t] = 0u;
}

__global__ void k_zero_f32(float* p, int n) {
  int t = blockIdx.x * blockDim.x + threadIdx.x;
  if (t < n) p[t] = 0.f;
}

__global__ void k_cvt(const float* __restrict__ s, h16* __restrict__ d,
                      int rows, int cols, int sld, int dld, int dcol) {
  int t = blockIdx.x * blockDim.x + threadIdx.x;
  if (t >= rows * cols) return;
  int r = t / cols, c = t % cols;
  d[(long long)r * dld + dcol + c] = (h16)s[(long long)r * sld + c];
}

// dst[c][r] = src[r][c]  (dst gets the transpose; dld = dst row length)
__global__ void k_cvtT(const float* __restrict__ s, h16* __restrict__ d,
                       int rows, int cols, int sld, int dld) {
  int t = blockIdx.x * blockDim.x + threadIdx.x;
  if (t >= rows * cols) return;
  int r = t / cols, c = t % cols;
  d[(long long)c * dld + r] = (h16)s[(long long)r * sld + c];
}

// ent_emb[b,e,:] = logsumexp over masked mentions of seq rows (OFFSET=1)
__global__ __launch_bounds__(256)
void k_ent_emb(const float* __restrict__ seq, const int* __restrict__ starts,
               const unsigned char* __restrict__ mask, float* __restrict__ ent)
{
  int b = blockIdx.x / NENT, e = blockIdx.x % NENT;
  int base = (b * NENT + e) * NMEN;
  int pos[NMEN]; float mk[NMEN]; float cnt = 0.f;
  for (int m = 0; m < NMEN; m++) {
    pos[m] = starts[base + m] + 1;
    mk[m]  = mask[base + m] ? 1.f : 0.f;
    cnt   += mk[m];
  }
  const float* sb = seq + (long long)b * SEQL * HIDD;
  float* ob = ent + (long long)(b * NENT + e) * HIDD;
  for (int d = threadIdx.x; d < HIDD; d += 256) {
    float mx = -1e30f;
    for (int m = 0; m < NMEN; m++) {
      float v = mk[m] > 0.f ? sb[(long long)pos[m] * HIDD + d] : -1e30f;
      mx = fmaxf(mx, v);
    }
    float ssum = 0.f;
    for (int m = 0; m < NMEN; m++) {
      float v = mk[m] > 0.f ? sb[(long long)pos[m] * HIDD + d] : -1e30f;
      ssum += expf(v - mx);
    }
    float lse = mx + logf(ssum);
    ob[d] = (cnt > 0.f) ? lse : 0.f;
  }
}

// ent_att[b,e,h,:] = mean over masked mentions of attention rows
__global__ __launch_bounds__(256)
void k_ent_att(const float* __restrict__ att, const int* __restrict__ starts,
               const unsigned char* __restrict__ mask, float* __restrict__ ea)
{
  int idx = blockIdx.x;
  int h = idx % NHEADS; int be = idx / NHEADS;
  int b = be / NENT, e = be % NENT;
  int base = (b * NENT + e) * NMEN;
  int pos[NMEN]; float mk[NMEN]; float cnt = 0.f;
  for (int m = 0; m < NMEN; m++) {
    pos[m] = starts[base + m] + 1;
    mk[m]  = mask[base + m] ? 1.f : 0.f;
    cnt   += mk[m];
  }
  const float* ab = att + ((long long)b * NHEADS + h) * SEQL * SEQL;
  float* ob = ea + ((long long)(b * NENT + e) * NHEADS + h) * SEQL;
  float inv = 1.f / (cnt + 1e-9f);
  for (int q = threadIdx.x; q < SEQL; q += 256) {
    float s = 0.f;
    for (int m = 0; m < NMEN; m++)
      if (mk[m] > 0.f) s += ab[(long long)pos[m] * SEQL + q];
    ob[q] = s * inv;
  }
}

// ht row (normalized product of entity attentions), stored f16 into padded rows
__global__ __launch_bounds__(256)
void k_ht(const float* __restrict__ ea, const int* __restrict__ hts, h16* __restrict__ ht16)
{
  __shared__ float red[256];
  int pr = blockIdx.x;
  int b  = pr / NPAIRS;
  int p  = pr % NPAIRS;
  int hi = hts[pr * 2 + 0], ti = hts[pr * 2 + 1];
  const float* eh = ea + (long long)(b * NENT + hi) * NHEADS * SEQL;
  const float* et = ea + (long long)(b * NENT + ti) * NHEADS * SEQL;
  float loc[4]; float mysum = 0.f;
  for (int qi = 0; qi < 4; qi++) {
    int q = threadIdx.x + qi * 256;
    float a = 0.f;
    for (int h = 0; h < NHEADS; h++) a += eh[h * SEQL + q] * et[h * SEQL + q];
    a *= (1.f / NHEADS);
    loc[qi] = a; mysum += a;
  }
  red[threadIdx.x] = mysum; __syncthreads();
  for (int s = 128; s > 0; s >>= 1) {
    if (threadIdx.x < s) red[threadIdx.x] += red[threadIdx.x + s];
    __syncthreads();
  }
  float inv = 1.f / (red[0] + 1e-5f);
  h16* ob = ht16 + (long long)(b * 1024 + p) * SEQL;   // padded 1024 rows/doc
  for (int qi = 0; qi < 4; qi++) {
    int q = threadIdx.x + qi * 256;
    ob[q] = (h16)(loc[qi] * inv);
  }
}

// gather hs/ts into f16 concat buffers (cols [0,768) of 1536-wide rows)
__global__ void k_gather_hts(const float* __restrict__ ent, const int* __restrict__ hts,
                             h16* __restrict__ cat_h, h16* __restrict__ cat_t)
{
  int t = blockIdx.x * blockDim.x + threadIdx.x;
  if (t >= TOTPAIR * HIDD) return;
  int pr = t / HIDD, d = t % HIDD;
  int b  = pr / NPAIRS;
  int hi = hts[pr * 2], ti = hts[pr * 2 + 1];
  cat_h[(long long)pr * (2 * HIDD) + d] = (h16)ent[(long long)(b * NENT + hi) * HIDD + d];
  cat_t[(long long)pr * (2 * HIDD) + d] = (h16)ent[(long long)(b * NENT + ti) * HIDD + d];
}

// ---- GAT pieces -----------------------------------------------------------
__global__ __launch_bounds__(128)
void k_el_er(const float* __restrict__ feat, const float* __restrict__ al,
             const float* __restrict__ ar, float* __restrict__ el,
             float* __restrict__ er, int heads, int outdim)
{
  __shared__ float r1[128], r2[128];
  int node = blockIdx.x / heads, h = blockIdx.x % heads;
  const float* f = feat + (long long)(node * heads + h) * outdim;
  float s1 = 0.f, s2 = 0.f;
  for (int d = threadIdx.x; d < outdim; d += 128) {
    float fv = f[d];
    s1 += fv * al[h * outdim + d];
    s2 += fv * ar[h * outdim + d];
  }
  r1[threadIdx.x] = s1; r2[threadIdx.x] = s2; __syncthreads();
  for (int s = 64; s > 0; s >>= 1) {
    if (threadIdx.x < s) { r1[threadIdx.x] += r1[threadIdx.x + s]; r2[threadIdx.x] += r2[threadIdx.x + s]; }
    __syncthreads();
  }
  if (!threadIdx.x) { el[node * heads + h] = r1[0]; er[node * heads + h] = r2[0]; }
}

__global__ void k_edge_e(const float* __restrict__ el, const float* __restrict__ er,
                         const int* __restrict__ src, const int* __restrict__ dst,
                         float* __restrict__ ebuf, int heads)
{
  int t = blockIdx.x * blockDim.x + threadIdx.x;
  if (t >= NEDGES * heads) return;
  int edge = t / heads, h = t % heads;
  float x = el[src[edge] * heads + h] + er[dst[edge] * heads + h];
  ebuf[t] = x > 0.f ? x : 0.2f * x;   // leaky_relu 0.2
}

__global__ __launch_bounds__(128)
void k_segmax(const float* __restrict__ ebuf, const int* __restrict__ dst,
              float* __restrict__ emax, int heads)
{
  __shared__ float red[128];
  int node = blockIdx.x / heads, h = blockIdx.x % heads;
  float mx = -1e30f;
  for (int e = threadIdx.x; e < NEDGES; e += 128)
    if (dst[e] == node) mx = fmaxf(mx, ebuf[e * heads + h]);
  red[threadIdx.x] = mx; __syncthreads();
  for (int s = 64; s > 0; s >>= 1) {
    if (threadIdx.x < s) red[threadIdx.x] = fmaxf(red[threadIdx.x], red[threadIdx.x + s]);
    __syncthreads();
  }
  if (!threadIdx.x) emax[node * heads + h] = red[0];
}

__global__ void k_edge_w(const float* __restrict__ ebuf, const float* __restrict__ emax,
                         const int* __restrict__ dst, float* __restrict__ wbuf,
                         float* __restrict__ den, int heads)
{
  int t = blockIdx.x * blockDim.x + threadIdx.x;
  if (t >= NEDGES * heads) return;
  int edge = t / heads, h = t % heads;
  float w = expf(ebuf[t] - emax[dst[edge] * heads + h]);
  wbuf[t] = w;
  atomicAdd(&den[dst[edge] * heads + h], w);
}

__global__ __launch_bounds__(256)
void k_scatter(const float* __restrict__ wbuf, const float* __restrict__ den,
               const float* __restrict__ feat, const int* __restrict__ src,
               const int* __restrict__ dst, float* __restrict__ gout,
               int heads, int outdim)
{
  int edge = blockIdx.x;
  int s = src[edge], dd = dst[edge];
  for (int h = 0; h < heads; h++) {
    float alpha = wbuf[edge * heads + h] / (den[dd * heads + h] + 1e-9f);
    const float* fs = feat + (long long)(s * heads + h) * outdim;
    float* od = gout + (long long)(dd * heads + h) * outdim;
    for (int d = threadIdx.x; d < outdim; d += 256)
      atomicAdd(&od[d], alpha * fs[d]);
  }
}

// h0 = elu(gout + b), written f16 into padded-lda buffer
__global__ void k_bias_elu_to_h16(const float* __restrict__ g, const float* __restrict__ b,
                                  h16* __restrict__ d, int n, int cols, int dld)
{
  int t = blockIdx.x * blockDim.x + threadIdx.x;
  if (t >= n) return;
  float v = g[t] + b[t % cols];
  v = v > 0.f ? v : (expf(v) - 1.f);   // elu
  d[(long long)(t / cols) * dld + (t % cols)] = (h16)v;
}

__global__ void k_bias_f32(const float* __restrict__ g, const float* __restrict__ b,
                           float* __restrict__ d, int n, int cols)
{
  int t = blockIdx.x * blockDim.x + threadIdx.x;
  if (t >= n) return;
  d[t] = g[t] + b[t % cols];
}

// layernorm over 97 labels; writes f16 into concat buffer at column offset
__global__ __launch_bounds__(128)
void k_layernorm(const float* __restrict__ src, h16* __restrict__ dst,
                 const float* __restrict__ g, const float* __restrict__ b,
                 int dld, int dcol)
{
  __shared__ float red[128];
  int r = blockIdx.x, tid = threadIdx.x;
  float x = (tid < NLAB) ? src[(long long)r * NLAB + tid] : 0.f;
  red[tid] = x; __syncthreads();
  for (int s = 64; s > 0; s >>= 1) {
    if (tid < s) red[tid] += red[tid + s];
    __syncthreads();
  }
  float mu = red[0] / NLAB;
  __syncthreads();
  float dv = (tid < NLAB) ? (x - mu) : 0.f;
  red[tid] = dv * dv; __syncthreads();
  for (int s = 64; s > 0; s >>= 1) {
    if (tid < s) red[tid] += red[tid + s];
    __syncthreads();
  }
  float var = red[0] / NLAB;
  if (tid < NLAB) {
    float y = (x - mu) * rsqrtf(var + 1e-5f) * g[tid] + b[tid];
    dst[(long long)r * dld + dcol + tid] = (h16)y;
  }
}

// ---------------------------------------------------------------------------
extern "C" void kernel_launch(void* const* d_in, const int* in_sizes, int n_in,
                              void* d_out, int out_size, void* d_ws, size_t ws_size,
                              hipStream_t stream)
{
  (void)in_sizes; (void)n_in; (void)out_size; (void)ws_size;
  const float* seq       = (const float*)d_in[0];
  const float* att       = (const float*)d_in[1];
  const int*   mstart    = (const int*)d_in[2];
  const unsigned char* mmask = (const unsigned char*)d_in[3];
  const int*   hts       = (const int*)d_in[4];
  const float* label_emb = (const float*)d_in[5];
  const float* gat_W0    = (const float*)d_in[6];
  const float* gat_al0   = (const float*)d_in[7];
  const float* gat_ar0   = (const float*)d_in[8];
  const float* gat_b0    = (const float*)d_in[9];
  const float* gat_W1    = (const float*)d_in[10];
  const float* gat_al1   = (const float*)d_in[11];
  const float* gat_ar1   = (const float*)d_in[12];
  const float* gat_b1    = (const float*)d_in[13];
  const int*   edge_src  = (const int*)d_in[14];
  const int*   edge_dst  = (const int*)d_in[15];
  const float* head_W    = (const float*)d_in[16];
  const float* head_b    = (const float*)d_in[17];
  const float* tail_W    = (const float*)d_in[18];
  const float* tail_b    = (const float*)d_in[19];
  const float* ln_g      = (const float*)d_in[20];
  const float* ln_b      = (const float*)d_in[21];
  const float* lin2_W    = (const float*)d_in[22];
  const float* lin2_b    = (const float*)d_in[23];
  const float* bil_W     = (const float*)d_in[24];
  const float* bil_b     = (const float*)d_in[25];
  float* out = (float*)d_out;

  // ---- workspace carve (256B aligned); all f16 buffers first (bulk-zeroed) --
  char* ws = (char*)d_ws;
  size_t off = 0;
  auto alloc = [&](size_t bytes) -> char* {
    char* p = ws + off;
    off += (bytes + 255) & ~(size_t)255;
    return p;
  };
  // padded f16 operands (row counts padded to grid coverage, K to 32-mult)
  h16* seqT16   = (h16*)alloc((size_t)N_DOCS * HIDD * SEQL * 2);   // [768][1024] per doc
  h16* ht16     = (h16*)alloc((size_t)N_DOCS * 1024 * SEQL * 2);   // [1024][1024] per doc
  h16* cat_h    = (h16*)alloc((size_t)PAIRPAD * 2 * HIDD * 2);     // [4096][1536]
  h16* cat_t    = (h16*)alloc((size_t)PAIRPAD * 2 * HIDD * 2);
  h16* label16  = (h16*)alloc((size_t)128 * EMBD * 2);             // [128][768]
  h16* W0T      = (h16*)alloc((size_t)1024 * EMBD * 2);            // [1024][768]
  h16* h0_16    = (h16*)alloc((size_t)128 * 1024 * 2);             // [128][1024]
  h16* W1T      = (h16*)alloc((size_t)EMBD * 1024 * 2);            // [768][1024]
  h16* lab16    = (h16*)alloc((size_t)128 * EMBD * 2);             // [128][768]  (BT for hs@lab.T)
  h16* headWT   = (h16*)alloc((size_t)EMBD * 2 * HIDD * 2);        // [768][1536]
  h16* tailWT   = (h16*)alloc((size_t)EMBD * 2 * HIDD * 2);
  h16* cat_l    = (h16*)alloc((size_t)PAIRPAD * 224 * 2);          // [4096][224] (K=194 padded)
  h16* lin2WT   = (h16*)alloc((size_t)128 * 224 * 2);              // [128][224]
  h16* ll16     = (h16*)alloc((size_t)PAIRPAD * 128 * 2);          // [4096][128] (K=97 padded)
  h16* bilWT    = (h16*)alloc((size_t)112 * KBIL * 2);             // [112][49152]
  h16* bilWtT   = (h16*)alloc((size_t)128 * 128 * 2);              // [128][128] tail BT
  size_t h16_bytes = off;                                          // bulk zero region

  float* ent_emb = (float*)alloc((size_t)N_DOCS * NENT * HIDD * 4);
  float* ent_att = (float*)alloc((size_t)N_DOCS * NENT * NHEADS * SEQL * 4);
  float* rs      = (float*)alloc((size_t)TOTPAIR * HIDD * 4);
  float* h2      = (float*)alloc((size_t)TOTPAIR * EMBD * 4);
  float* t2      = (float*)alloc((size_t)TOTPAIR * EMBD * 4);
  float* feat0   = (float*)alloc((size_t)NLAB * 2 * GHIDD * 4);
  float* gout0   = (float*)alloc((size_t)NLAB * 2 * GHIDD * 4);
  float* feat1   = (float*)alloc((size_t)NLAB * EMBD * 4);
  float* gout1   = (float*)alloc((size_t)NLAB * EMBD * 4);
  float* lab_f32 = (float*)alloc((size_t)NLAB * EMBD * 4);
  float* hsl_raw = (float*)alloc((size_t)TOTPAIR * NLAB * 4);
  float* tsl_raw = (float*)alloc((size_t)TOTPAIR * NLAB * 4);
  float* ll_f32  = (float*)alloc((size_t)TOTPAIR * NLAB * 4);
  float* el0  = (float*)alloc(NLAB * 2 * 4);
  float* er0  = (float*)alloc(NLAB * 2 * 4);
  float* e0   = (float*)alloc(NEDGES * 2 * 4);
  float* w0   = (float*)alloc(NEDGES * 2 * 4);
  float* em0  = (float*)alloc(NLAB * 2 * 4);
  float* den0 = (float*)alloc(NLAB * 2 * 4);
  float* el1  = (float*)alloc(NLAB * 4);
  float* er1  = (float*)alloc(NLAB * 4);
  float* e1   = (float*)alloc(NEDGES * 4);
  float* w1   = (float*)alloc(NEDGES * 4);
  float* em1  = (float*)alloc(NLAB * 4);
  float* den1 = (float*)alloc(NLAB * 4);

  // ---- bulk zero of all padded f16 staging (keeps pads zero every replay) --
  {
    long long nw = (long long)(h16_bytes / 4);
    k_zero_u32<<<(unsigned)((nw + 255) / 256), 256, 0, stream>>>((unsigned int*)ws, nw);
  }

  auto cvt = [&](const float* s, h16* d, int rows, int cols, int sld, int dld, int dcol) {
    int n = rows * cols;
    k_cvt<<<(n + 255) / 256, 256, 0, stream>>>(s, d, rows, cols, sld, dld, dcol);
  };
  auto cvtT = [&](const float* s, h16* d, int rows, int cols, int sld, int dld) {
    int n = rows * cols;
    k_cvtT<<<(n + 255) / 256, 256, 0, stream>>>(s, d, rows, cols, sld, dld);
  };
  auto gemm = [&](const h16* A, const h16* BT, float* C, const float* bias,
                  int M, int N, int Kp, int lda, int ldb, int ldc,
                  long long sa, long long sb, long long sc, int batch, int epi) {
    dim3 g((N + 63) / 64, (M + 63) / 64, batch);
    k_gemm<<<g, 128, 0, stream>>>(A, BT, C, bias, M, N, Kp, lda, ldb, ldc, sa, sb, sc, epi);
  };

  // ---- f16 staging (A row-major, B transposed) ----
  for (int z = 0; z < N_DOCS; z++)
    cvtT(seq + (long long)z * SEQL * HIDD, seqT16 + (long long)z * HIDD * SEQL,
         SEQL, HIDD, HIDD, SEQL);                                  // [768][1024]
  cvt (label_emb, label16, NLAB, EMBD, EMBD, EMBD, 0);
  cvtT(gat_W0, W0T, EMBD, 2 * GHIDD, 2 * GHIDD, EMBD);             // [1000->1024][768]
  cvtT(gat_W1, W1T, 2 * GHIDD, EMBD, EMBD, 1024);                  // [768][1000->1024]
  cvtT(head_W, headWT, 2 * HIDD, EMBD, EMBD, 2 * HIDD);            // [768][1536]
  cvtT(tail_W, tailWT, 2 * HIDD, EMBD, EMBD, 2 * HIDD);
  cvtT(lin2_W, lin2WT, 2 * NLAB, NLAB, NLAB, 224);                 // [97->128][194->224]
  cvtT(bil_W, bilWT, KBIL, NLAB, NLAB, KBIL);                      // [97->112][49152]
  cvtT(bil_W + (long long)KBIL * NLAB, bilWtT, NLAB, NLAB, NLAB, 128); // [97->128][97->128]

  // ---- entity pooling ----
  k_ent_emb<<<N_DOCS * NENT, 256, 0, stream>>>(seq, mstart, mmask, ent_emb);
  k_ent_att<<<N_DOCS * NENT * NHEADS, 256, 0, stream>>>(att, mstart, mmask, ent_att);
  k_ht<<<TOTPAIR, 256, 0, stream>>>(ent_att, hts, ht16);
  k_gather_hts<<<(TOTPAIR * HIDD + 255) / 256, 256, 0, stream>>>(ent_emb, hts, cat_h, cat_t);

  // ---- rs = ht @ seq (batched over docs) ----
  gemm(ht16, seqT16, rs, nullptr, NPAIRS, HIDD, SEQL, SEQL, SEQL, HIDD,
       (long long)1024 * SEQL, (long long)HIDD * SEQL, (long long)NPAIRS * HIDD, N_DOCS, 0);
  cvt(rs, cat_h, TOTPAIR, HIDD, HIDD, 2 * HIDD, HIDD);
  cvt(rs, cat_t, TOTPAIR, HIDD, HIDD, 2 * HIDD, HIDD);

  // ---- GAT layer 0 (heads=2, out=500) ----
  gemm(label16, W0T, feat0, nullptr, NLAB, 2 * GHIDD, EMBD, EMBD, EMBD, 2 * GHIDD, 0, 0, 0, 1, 0);
  k_el_er<<<NLAB * 2, 128, 0, stream>>>(feat0, gat_al0, gat_ar0, el0, er0, 2, GHIDD);
  k_edge_e<<<(NEDGES * 2 + 127) / 128, 128, 0, stream>>>(el0, er0, edge_src, edge_dst, e0, 2);
  k_segmax<<<NLAB * 2, 128, 0, stream>>>(e0, edge_dst, em0, 2);
  k_zero_f32<<<1, 256, 0, stream>>>(den0, NLAB * 2);
  k_edge_w<<<(NEDGES * 2 + 127) / 128, 128, 0, stream>>>(e0, em0, edge_dst, w0, den0, 2);
  k_zero_f32<<<(NLAB * 2 * GHIDD + 255) / 256, 256, 0, stream>>>(gout0, NLAB * 2 * GHIDD);
  k_scatter<<<NEDGES, 256, 0, stream>>>(w0, den0, feat0, edge_src, edge_dst, gout0, 2, GHIDD);
  k_bias_elu_to_h16<<<(NLAB * 2 * GHIDD + 255) / 256, 256, 0, stream>>>(
      gout0, gat_b0, h0_16, NLAB * 2 * GHIDD, 2 * GHIDD, 1024);

  // ---- GAT layer 1 (heads=1, out=768) ----
  gemm(h0_16, W1T, feat1, nullptr, NLAB, EMBD, 1024, 1024, 1024, EMBD, 0, 0, 0, 1, 0);
  k_el_er<<<NLAB, 128, 0, stream>>>(feat1, gat_al1, gat_ar1, el1, er1, 1, EMBD);
  k_edge_e<<<(NEDGES + 127) / 128, 128, 0, stream>>>(el1, er1, edge_src, edge_dst, e1, 1);
  k_segmax<<<NLAB, 128, 0, stream>>>(e1, edge_dst, em1, 1);
  k_zero_f32<<<1, 256, 0, stream>>>(den1, NLAB);
  k_edge_w<<<(NEDGES + 127) / 128, 128, 0, stream>>>(e1, em1, edge_dst, w1, den1, 1);
  k_zero_f32<<<(NLAB * EMBD + 255) / 256, 256, 0, stream>>>(gout1, NLAB * EMBD);
  k_scatter<<<NEDGES, 256, 0, stream>>>(w1, den1, feat1, edge_src, edge_dst, gout1, 1, EMBD);
  k_bias_f32<<<(NLAB * EMBD + 255) / 256, 256, 0, stream>>>(gout1, gat_b1, lab_f32, NLAB * EMBD, EMBD);
  cvt(lab_f32, lab16, NLAB, EMBD, EMBD, EMBD, 0);   // lab itself is BT of lab.T

  // ---- hsl/tsl = layernorm(hs @ lab.T) -> cat_l ----
  gemm(cat_h, lab16, hsl_raw, nullptr, TOTPAIR, NLAB, EMBD, 2 * HIDD, EMBD, NLAB, 0, 0, 0, 1, 0);
  gemm(cat_t, lab16, tsl_raw, nullptr, TOTPAIR, NLAB, EMBD, 2 * HIDD, EMBD, NLAB, 0, 0, 0, 1, 0);
  k_layernorm<<<TOTPAIR, 128, 0, stream>>>(hsl_raw, cat_l, ln_g, ln_b, 224, 0);
  k_layernorm<<<TOTPAIR, 128, 0, stream>>>(tsl_raw, cat_l, ln_g, ln_b, 224, NLAB);

  // ---- h2/t2 = tanh(concat @ W + b) ----
  gemm(cat_h, headWT, h2, head_b, TOTPAIR, EMBD, 2 * HIDD, 2 * HIDD, 2 * HIDD, EMBD, 0, 0, 0, 1, 2);
  gemm(cat_t, tailWT, t2, tail_b, TOTPAIR, EMBD, 2 * HIDD, 2 * HIDD, 2 * HIDD, EMBD, 0, 0, 0, 1, 2);

  // ---- logits_lab = cat_l @ lin2_W + b ----
  gemm(cat_l, lin2WT, ll_f32, lin2_b, TOTPAIR, NLAB, 224, 224, 224, NLAB, 0, 0, 0, 1, 1);
  cvt(ll_f32, ll16, TOTPAIR, NLAB, NLAB, 128, 0);

  // ---- bilinear head -> out, then out += logits_lab @ bil_W[49152:] ----
  k_bilinear<<<TOTPAIR / 16, 224, 0, stream>>>(h2, t2, bilWT, bil_b, out);
  gemm(ll16, bilWtT, out, nullptr, TOTPAIR, NLAB, 128, 128, 128, NLAB, 0, 0, 0, 1, 3);
}
